// Attention_aggregator_85899346055
// MI455X (gfx1250) — matrix operand order repaired and checked by reading, not compiled, not run on hardware
//
#include <hip/hip_runtime.h>
#include <hip/hip_bf16.h>

// ---------------------------------------------------------------------------
// Fused graph-attention aggregator for MI455X (gfx1250, wave32, WMMA + TDM).
//   S = X Xt (masked by adj + self loops), P = softmax_rows(S), O = P X
// One block = 256 threads = 8 wave32, owns a 16-query tile of one batch.
// Whole 16x2048 score row-block lives in LDS (320KB/WGP) -> no flash rescale.
// Both GEMMs use v_wmma_f32_16x16x32_bf16; adjacency tiles arrive via the
// Tensor Data Mover (tensor_load_to_lds + s_wait_tensorcnt).
// A-fragments are re-read from LDS each K-step: keeping them in registers
// spills to scratch (verified in round-2 asm), and ds_load_b128 co-issues
// cleanly with WMMA.
// ---------------------------------------------------------------------------

typedef __bf16 bf16_t;
typedef bf16_t v16bf __attribute__((ext_vector_type(16)));
typedef bf16_t v8bf  __attribute__((ext_vector_type(8)));
typedef float  v8f   __attribute__((ext_vector_type(8)));
typedef unsigned int v4u __attribute__((ext_vector_type(4)));
typedef int    v8i   __attribute__((ext_vector_type(8)));
typedef int    v4i   __attribute__((ext_vector_type(4)));

#define B_  8
#define N_  2048
#define D_  512
#define TQ  16            // query rows per block
#define CK  128           // keys staged per LDS chunk
#define NCHUNK (N_ / CK)  // 16
#define NEG_INF_F (-9999999.0f)

// LDS strides (in elements), padded so every vector load is 32B aligned.
#define SQ_STRIDE  528    // bf16: row = 1056 B (32*33)
#define SS_STRIDE  2056   // f32 : row = 8224 B (32*257)
#define SP_STRIDE  4112   // bf16 view of the same rows (8224 B)
#define SKV_STRIDE 528    // bf16 key staging rows
#define SVT_STRIDE 144    // bf16 transposed V rows: row = 288 B (32*9)

// dynamic-LDS byte offsets
#define SQ_OFF    0
#define SS_OFF    (SQ_OFF + TQ * SQ_STRIDE * 2)          // 16896
#define SKV_OFF   (SS_OFF + TQ * SS_STRIDE * 4)          // 148480
#define SKV_BYTES (CK * SKV_STRIDE * 2)                  // 135168 (phase 1)
#define SVT_BYTES (D_ * SVT_STRIDE * 2)                  // 147456 (phase 3)
#define ADJ_OFF   (SKV_OFF + (SVT_BYTES > SKV_BYTES ? SVT_BYTES : SKV_BYTES)) // 295936
#define LDS_BYTES (ADJ_OFF + TQ * CK * 4)                // 304128 (< 320KB)

__device__ __forceinline__ bf16_t f2bf(float f) {
  unsigned u = __builtin_bit_cast(unsigned, f);
  u += 0x7FFFu + ((u >> 16) & 1u);              // round-to-nearest-even
  unsigned short h = (unsigned short)(u >> 16);
  return __builtin_bit_cast(bf16_t, h);
}

// Issue a TDM 2D tile load: 16 rows x CK int32 columns of the adjacency
// matrix -> LDS at byte offset ADJ_OFF.  D# layout per CDNA5 ISA ch.8.
// amdgpu-toolchain (clang-23) builtin form: 6 args
//   (v4u group0, v8i group1, v4i group2, v4i group3, v8i extra, i32 cpol)
__device__ __forceinline__ void tdm_load_adj_tile(const int* gsrc) {
  unsigned long long ga = (unsigned long long)(uintptr_t)gsrc;
  v4u g0;
  g0.x = 1u;                                    // count=1, is_restore=0
  g0.y = (unsigned)ADJ_OFF;                     // lds_addr (bytes)
  g0.z = (unsigned)(ga & 0xFFFFFFFFu);          // global_addr[31:0]
  g0.w = (unsigned)((ga >> 32) & 0x01FFFFFFu)   // global_addr[56:32]
       | (2u << 30);                            // type = 2 ("image")
  v8i g1;
  g1[0] = (int)(2u << 16);                      // data_size=4B; no mask/pad
  g1[1] = (int)((unsigned)N_ << 16);            // tensor_dim0 = 2048 (lo16)
  g1[2] = (int)((unsigned)N_ << 16);            // dim0 hi16=0 | tensor_dim1 lo16
  g1[3] = (int)((unsigned)CK << 16);            // dim1 hi16=0 | tile_dim0 = 128
  g1[4] = TQ;                                   // tile_dim1 = 16 (tile_dim2=0)
  g1[5] = N_;                                   // tensor_dim0_stride = 2048
  g1[6] = 0;                                    // stride0 hi | stride1 lo (unused 2D)
  g1[7] = 0;
  v4i gz4 = {0, 0, 0, 0};                       // groups 2/3 unused (2D tile)
  v8i gz8 = {0, 0, 0, 0, 0, 0, 0, 0};
  __builtin_amdgcn_tensor_load_to_lds(g0, g1, gz4, gz4, gz8, 0);
}

__global__ __launch_bounds__(256, 1)
void gat_fused_kernel(const float* __restrict__ X,
                      const int*   __restrict__ adj,
                      float*       __restrict__ out)
{
  extern __shared__ char smem[];
  bf16_t* sQ   = (bf16_t*)(smem + SQ_OFF);   // [TQ][SQ_STRIDE]  bf16 query tile
  float*  sS   = (float*) (smem + SS_OFF);   // [TQ][SS_STRIDE]  f32 scores
  bf16_t* sP   = (bf16_t*)(smem + SS_OFF);   // same rows, bf16 probs (in-place)
  bf16_t* sKV  = (bf16_t*)(smem + SKV_OFF);  // [CK][SKV_STRIDE] key chunk
  bf16_t* sVT  = (bf16_t*)(smem + SKV_OFF);  // [D_][SVT_STRIDE] transposed V chunk
  int*    sAdj = (int*)   (smem + ADJ_OFF);  // [TQ][CK] adjacency tile (TDM dest)

  const int tid  = threadIdx.x;
  const int wave = tid >> 5;
  const int lane = tid & 31;
  const int half = lane >> 4;     // WMMA lane half
  const int l16  = lane & 15;     // A-row / B-col / C-col lane index

  const int qt = blockIdx.x % (N_ / TQ);
  const int b  = blockIdx.x / (N_ / TQ);
  const int q0 = qt * TQ;

  const float* Xb = X   + (size_t)b * N_ * D_;
  const int*   Ab = adj + (size_t)b * N_ * N_;

  // ---------------- Phase 0: stage Q tile (16 x 512) as bf16 ----------------
  for (int i = tid; i < TQ * D_; i += 256) {
    int r = i >> 9, d = i & (D_ - 1);
    sQ[r * SQ_STRIDE + d] = f2bf(Xb[(size_t)(q0 + r) * D_ + d]);
  }

  // ---------------- Phase 1: S = Q K^T, masked, f32 into LDS ----------------
  for (int kc = 0; kc < NCHUNK; ++kc) {
    __syncthreads();                              // protect sKV / sAdj reuse
    // TDM: DMA this chunk's 16x128 adjacency tile into LDS (one wave issues;
    // overlaps with the bf16 staging loop below).
    if (wave == 0) {
      tdm_load_adj_tile(Ab + (size_t)q0 * N_ + kc * CK);
    }
    // Prefetch next chunk's key rows into L2 while we work on this one.
    if (kc + 1 < NCHUNK) {
      const float* Kn = Xb + (size_t)(kc + 1) * CK * D_;
      __builtin_prefetch((const void*)(Kn + (size_t)tid * 256), 0, 2);
    }
    // Stage 128 keys x 512 d as bf16 (coalesced read + convert).
    const float* Kg = Xb + (size_t)kc * CK * D_;
    for (int i = tid; i < CK * D_; i += 256) {
      int r = i >> 9, d = i & (D_ - 1);
      sKV[r * SKV_STRIDE + d] = f2bf(Kg[(size_t)r * D_ + d]);
    }
    if (wave == 0) {
      __builtin_amdgcn_s_wait_tensorcnt(0);       // adjacency tile landed
    }
    __syncthreads();

    v8f c = {};
    const int klocal = wave * 16;                 // this wave's 16-key tile
    #pragma unroll 4
    for (int dk = 0; dk < D_ / 32; ++dk) {        // 16 K-steps over D=512
      // A fragment (Q 16x32): lane l16 = query row; K = 16*half .. 16*half+15
      const bf16_t* arow = sQ + l16 * SQ_STRIDE + dk * 32;
      v8bf a0 = *(const v8bf*)(arow + 16 * half);
      v8bf a1 = *(const v8bf*)(arow + 16 * half + 8);
      v16bf a;
      #pragma unroll
      for (int e = 0; e < 8; ++e) { a[e] = a0[e]; a[8 + e] = a1[e]; }
      // B fragment (K^T 32x16): lane l16 = key; elements = 16 contiguous d
      const bf16_t* brow = sKV + (klocal + l16) * SKV_STRIDE + dk * 32 + 16 * half;
      v16bf bb = *(const v16bf*)brow;
      c = __builtin_amdgcn_wmma_f32_16x16x32_bf16(false, a, false, bb,
                                                  (short)0, c, false, false);
    }

    // mask (adj + self-loop) from the TDM-loaded LDS tile, spill f32 scores
    const int keyl = klocal + l16;                // local key in chunk
    const int key  = kc * CK + keyl;              // C: N = lane&15
    #pragma unroll
    for (int r = 0; r < 8; ++r) {
      int qrow = r + 8 * half;                    // C: M = r + 8*half
      int q    = q0 + qrow;
      int av   = sAdj[qrow * CK + keyl];
      float v  = ((av != 0) || (q == key)) ? c[r] : NEG_INF_F;
      sS[qrow * SS_STRIDE + key] = v;
    }
  }
  __syncthreads();

  // ---------------- Phase 2: row softmax, write bf16 P in place -------------
  for (int rr = 0; rr < 2; ++rr) {
    int row = wave * 2 + rr;                      // 8 waves x 2 rows = 16
    float* srow = sS + row * SS_STRIDE;
    float mx = -3.4e38f;
    for (int i = lane; i < N_; i += 32) mx = fmaxf(mx, srow[i]);
    #pragma unroll
    for (int s = 16; s >= 1; s >>= 1) mx = fmaxf(mx, __shfl_xor(mx, s, 32));
    float sum = 0.0f;
    for (int i = lane; i < N_; i += 32) sum += __expf(srow[i] - mx);
    #pragma unroll
    for (int s = 16; s >= 1; s >>= 1) sum += __shfl_xor(sum, s, 32);
    float inv = 1.0f / sum;                       // diagonal guarantees sum>0
    bf16_t* prow = sP + row * SP_STRIDE;          // bf16 i clobbers f32 i/2:
    for (int i = lane; i < N_; i += 32) {         // already-consumed (in-order DS)
      prow[i] = f2bf(__expf(srow[i] - mx) * inv);
    }
  }
  __syncthreads();

  // ---------------- Phase 3: O = P V, each wave owns 64 d-columns -----------
  v8f acc[4] = {};
  const int d0 = wave * 64;
  for (int kc = 0; kc < NCHUNK; ++kc) {
    __syncthreads();                              // protect sVT reuse
    if (kc + 1 < NCHUNK) {
      const float* Vn = Xb + (size_t)(kc + 1) * CK * D_;
      __builtin_prefetch((const void*)(Vn + (size_t)tid * 256), 0, 2);
    }
    const float* Vg = Xb + (size_t)kc * CK * D_;
    for (int i = tid; i < CK * D_; i += 256) {    // coalesced load, transposed store
      int r = i >> 9, d = i & (D_ - 1);
      sVT[d * SVT_STRIDE + r] = f2bf(Vg[(size_t)r * D_ + d]);
    }
    __syncthreads();

    #pragma unroll
    for (int st = 0; st < CK / 32; ++st) {        // 4 K-steps of 32 keys
      // A fragment (P 16x32): lane l16 = query row
      const bf16_t* prow = sP + l16 * SP_STRIDE + kc * CK + st * 32;
      v8bf a0 = *(const v8bf*)(prow + 16 * half);
      v8bf a1 = *(const v8bf*)(prow + 16 * half + 8);
      v16bf a;
      #pragma unroll
      for (int e = 0; e < 8; ++e) { a[e] = a0[e]; a[8 + e] = a1[e]; }
      #pragma unroll
      for (int t = 0; t < 4; ++t) {
        // B fragment (V 32x16): lane l16 = d-column; 16 contiguous keys
        const bf16_t* bcol = sVT + (size_t)(d0 + t * 16 + l16) * SVT_STRIDE
                                 + st * 32 + 16 * half;
        v16bf bb = *(const v16bf*)bcol;
        acc[t] = __builtin_amdgcn_wmma_f32_16x16x32_bf16(false, a, false, bb,
                                                         (short)0, acc[t],
                                                         false, false);
      }
    }
  }

  // ---------------- Epilogue: write O tile (f32) ----------------------------
  float* Ob = out + (size_t)b * N_ * D_;
  #pragma unroll
  for (int t = 0; t < 4; ++t) {
    #pragma unroll
    for (int r = 0; r < 8; ++r) {
      Ob[(size_t)(q0 + r + 8 * half) * D_ + d0 + t * 16 + l16] = acc[t][r];
    }
  }
}

extern "C" void kernel_launch(void* const* d_in, const int* in_sizes, int n_in,
                              void* d_out, int out_size, void* d_ws, size_t ws_size,
                              hipStream_t stream) {
  (void)in_sizes; (void)n_in; (void)out_size; (void)d_ws; (void)ws_size;
  const float* X   = (const float*)d_in[0];   // node_features [8,2048,512] f32
  // d_in[1] = nodes (unused by forward)
  const int*   adj = (const int*)d_in[2];     // adj_list [8,2048,2048] i32
  float* out = (float*)d_out;                 // [8,2048,512] f32

  dim3 grid(B_ * (N_ / TQ));                  // 1024 blocks
  dim3 block(256);                            // 8 wave32
  gat_fused_kernel<<<grid, block, LDS_BYTES, stream>>>(X, adj, out);
}
